// SyndromeTransformer_25151328485481
// MI455X (gfx1250) — matrix operand order repaired and checked by reading, not compiled, not run on hardware
//
#include <hip/hip_runtime.h>

typedef __bf16 bf16_t;
typedef __bf16 v16bf __attribute__((ext_vector_type(16)));
typedef float  v8f   __attribute__((ext_vector_type(8)));

// ---------------- workspace layout (bf16 element offsets) ----------------
#define WS_WEMB 0          // [256][32]  (K padded 16->32 with zeros)
#define WS_WQ   8192       // [64][256]
#define WS_WK   24576
#define WS_WV   40960
#define WS_WO   57344      // [256][64]
#define WS_W1   73728      // [4][128][256]
#define WS_W2   204800     // [4][256][128]
#define WS_CONV 335872     // [9][256][256]  (per-tap, [dout][din])
#define WS_BF16_END 925696 // pos_bias f32 follows at byte 2*WS_BF16_END

// conv edges regrouped by tap (taps 0,2,4,6,8 -> t2=tap/2):
// TAPSI[t2][s_out] = s_in, or -1 if no edge
__constant__ int TAPSI[5][8] = {
  {-1,-1,-1, 0, 2, 3,-1, 4},   // tap 0
  {-1,-1, 0, 1, 3,-1, 4, 5},   // tap 2
  { 0, 1, 2, 3, 4, 5, 6, 7},   // tap 4 (center)
  { 2, 3,-1, 4, 6, 7,-1,-1},   // tap 6
  { 3,-1, 4, 5, 7,-1,-1,-1},   // tap 8
};

struct P26 { const float* p[26]; };

// ---- register-blocked GEMM: 4 M-tiles x 2 N-tiles per wave ----
// A: row-major [64][K] bf16 (LDS), B0/B1: row-major [16][K] bf16 = B^T cols
__device__ inline void gemm_4x2(const bf16_t* A0, int lda,
                                const bf16_t* B0, const bf16_t* B1, int ldb,
                                int K, v8f c[4][2], int lane) {
  const int m = lane & 15, half = lane >> 4;
  for (int k0 = 0; k0 < K; k0 += 32) {
    union { v16bf v; uint4 q[2]; } b0, b1;
    const bf16_t* bp0 = B0 + m * ldb + k0 + half * 16;
    b0.q[0] = *(const uint4*)(bp0);
    b0.q[1] = *(const uint4*)(bp0 + 8);
    const bf16_t* bp1 = B1 + m * ldb + k0 + half * 16;
    b1.q[0] = *(const uint4*)(bp1);
    b1.q[1] = *(const uint4*)(bp1 + 8);
    #pragma unroll
    for (int mt = 0; mt < 4; ++mt) {
      union { v16bf v; uint4 q[2]; } a;
      const bf16_t* ap = A0 + (mt * 16 + m) * lda + k0 + half * 16;
      a.q[0] = *(const uint4*)(ap);
      a.q[1] = *(const uint4*)(ap + 8);
      c[mt][0] = __builtin_amdgcn_wmma_f32_16x16x32_bf16(false, a.v, false, b0.v,
                                                         (short)0, c[mt][0], false, false);
      c[mt][1] = __builtin_amdgcn_wmma_f32_16x16x32_bf16(false, a.v, false, b1.v,
                                                         (short)0, c[mt][1], false, false);
    }
  }
}

__device__ inline void zero_acc(v8f c[4][2]) {
  const v8f z = {0.f,0.f,0.f,0.f,0.f,0.f,0.f,0.f};
  #pragma unroll
  for (int i = 0; i < 4; ++i) { c[i][0] = z; c[i][1] = z; }
}

// 2 threads per row; writes normalized row back to hf+hb, or to global (gout)
__device__ inline void layernorm_rows(float* hf, bf16_t* hb, const float* sc,
                                      const float* of, int tid, float* gout) {
  const int row  = tid >> 1;
  const int part = tid & 1;
  float* base = hf + row * 260;
  const int j0 = part * 128;
  float s = 0.f, ss = 0.f;
  for (int j = j0; j < j0 + 128; ++j) { float v = base[j]; s += v; ss += v * v; }
  s  += __shfl_xor(s, 1);
  ss += __shfl_xor(ss, 1);
  float mean = s * (1.0f / 256.0f);
  float var  = ss * (1.0f / 256.0f) - mean * mean;
  float rstd = rsqrtf(var + 1e-5f);
  for (int j = j0; j < j0 + 128; ++j) {
    float v = (base[j] - mean) * rstd * sc[j] + of[j];
    if (gout) { gout[j] = v; }
    else      { base[j] = v; hb[row * 264 + j] = (bf16_t)v; }
  }
}

// ---------------- main fused kernel: 16 batches (128 rows) per WG ----------------
__global__ __launch_bounds__(256)
void fused_block(P26 in, const unsigned short* wsb, const float* pos_bias, float* out) {
  extern __shared__ char smem[];
  float*  hf = (float*)smem;                    // [128][260] f32 residual stream
  bf16_t* hb = (bf16_t*)(smem + 133120);        // [128][264] bf16 copy (WMMA A)
  bf16_t* ub = (bf16_t*)(smem + 200704);        // 73728B union: x / qkv+ctx / hid
  float* glog  = (float*)(smem + 274432);       // [128][4]
  float* gatev = (float*)(smem + 276480);       // [128][4]

  const bf16_t* ws = (const bf16_t*)wsb;
  const int tid = threadIdx.x, wid = tid >> 5, lane = tid & 31;
  const int nloc = lane & 15, half = lane >> 4;
  const int b0 = blockIdx.x * 16;

  // ---- Phase A: stage x as bf16, K padded to 32 ----
  bf16_t* xb = ub;                              // [128][40]
  {
    const float* x = in.p[0];
    for (int i = tid; i < 128 * 32; i += 256) {
      int r = i >> 5, f = i & 31;
      int lb = r & 15, s = r >> 4;
      float v = (f < 16) ? x[((b0 + lb) * 8 + s) * 16 + f] : 0.f;
      xb[r * 40 + f] = (bf16_t)v;
    }
  }
  __syncthreads();

  // ---- Phase B: h = x @ w_embed + b_embed ----
  {
    const float* be = in.p[2];
    for (int u = wid; u < 16; u += 8) {
      int nt2 = u & 7, mh = u >> 3;
      v8f c[4][2]; zero_acc(c);
      gemm_4x2(xb + mh * 64 * 40, 40,
               ws + WS_WEMB + (nt2 * 2 + 0) * 16 * 32,
               ws + WS_WEMB + (nt2 * 2 + 1) * 16 * 32, 32, 32, c, lane);
      #pragma unroll
      for (int n = 0; n < 2; ++n) {
        int col = (nt2 * 2 + n) * 16 + nloc;
        float bias = be[col];
        #pragma unroll
        for (int mt = 0; mt < 4; ++mt) {
          #pragma unroll
          for (int i = 0; i < 8; ++i) {
            int row = (mh * 4 + mt) * 16 + half * 8 + i;
            float v = c[mt][n][i] + bias;
            hf[row * 260 + col] = v;
            hb[row * 264 + col] = (bf16_t)v;
          }
        }
      }
    }
  }
  __syncthreads();

  // ---- Phase C: Q,K,V projections (q scaled by 1/sqrt(16)) ----
  bf16_t* qb  = ub;
  bf16_t* kbv = ub + 9216;
  bf16_t* vbv = ub + 18432;
  bf16_t* cxb = ub + 27648;
  for (int u = wid; u < 12; u += 8) {
    int pj = u >> 2, r = u & 3;
    int nt2 = r & 1, mh = r >> 1;
    const bf16_t* wt = ws + (pj == 0 ? WS_WQ : pj == 1 ? WS_WK : WS_WV);
    const float* bias = in.p[pj == 0 ? 4 : pj == 1 ? 6 : 8];
    bf16_t* dst = pj == 0 ? qb : pj == 1 ? kbv : vbv;
    float scl = (pj == 0) ? 0.25f : 1.0f;
    v8f c[4][2]; zero_acc(c);
    gemm_4x2(hb + mh * 64 * 264, 264,
             wt + (nt2 * 2 + 0) * 16 * 256,
             wt + (nt2 * 2 + 1) * 16 * 256, 256, 256, c, lane);
    #pragma unroll
    for (int n = 0; n < 2; ++n) {
      int col = (nt2 * 2 + n) * 16 + nloc;
      float bv = bias[col];
      #pragma unroll
      for (int mt = 0; mt < 4; ++mt) {
        #pragma unroll
        for (int i = 0; i < 8; ++i) {
          int row = (mh * 4 + mt) * 16 + half * 8 + i;
          dst[row * 72 + col] = (bf16_t)((c[mt][n][i] + bv) * scl);
        }
      }
    }
  }
  __syncthreads();

  // ---- Phase D: attention (S=8, per-head 8x8, VALU) ----
  for (int u = tid; u < 512; u += 256) {
    int sq = u & 7, hh = (u >> 3) & 3, lb = u >> 5;
    int rq = sq * 16 + lb;
    const bf16_t* qp = qb + rq * 72 + hh * 16;
    float qv[16];
    #pragma unroll
    for (int d = 0; d < 16; ++d) qv[d] = (float)qp[d];
    float lg[8], mx = -1e30f;
    #pragma unroll
    for (int sk = 0; sk < 8; ++sk) {
      const bf16_t* kp = kbv + (sk * 16 + lb) * 72 + hh * 16;
      float acc = 0.f;
      #pragma unroll
      for (int d = 0; d < 16; ++d) acc += qv[d] * (float)kp[d];
      lg[sk] = acc; mx = fmaxf(mx, acc);
    }
    float wsum = 0.f, w[8];
    #pragma unroll
    for (int sk = 0; sk < 8; ++sk) { w[sk] = __expf(lg[sk] - mx); wsum += w[sk]; }
    float inv = 1.f / wsum;
    bf16_t* cp = cxb + rq * 72 + hh * 16;
    #pragma unroll
    for (int d = 0; d < 16; ++d) {
      float a = 0.f;
      #pragma unroll
      for (int sk = 0; sk < 8; ++sk)
        a += w[sk] * (float)vbv[(sk * 16 + lb) * 72 + hh * 16 + d];
      cp[d] = (bf16_t)(a * inv);
    }
  }
  __syncthreads();

  // ---- Phase E: attn_out = ctx @ wo + bo ; residual into hf ----
  {
    const float* bo = in.p[10];
    for (int u = wid; u < 16; u += 8) {
      int nt2 = u & 7, mh = u >> 3;
      v8f c[4][2]; zero_acc(c);
      gemm_4x2(cxb + mh * 64 * 72, 72,
               ws + WS_WO + (nt2 * 2 + 0) * 16 * 64,
               ws + WS_WO + (nt2 * 2 + 1) * 16 * 64, 64, 64, c, lane);
      #pragma unroll
      for (int n = 0; n < 2; ++n) {
        int col = (nt2 * 2 + n) * 16 + nloc;
        float bv = bo[col];
        #pragma unroll
        for (int mt = 0; mt < 4; ++mt) {
          #pragma unroll
          for (int i = 0; i < 8; ++i) {
            int row = (mh * 4 + mt) * 16 + half * 8 + i;
            hf[row * 260 + col] += c[mt][n][i] + bv;
          }
        }
      }
    }
  }
  __syncthreads();

  // ---- Phase F: LN1 ----
  layernorm_rows(hf, hb, in.p[11], in.p[12], tid, nullptr);
  __syncthreads();

  // ---- Phase G: gate = softmax(h @ wg + bg) ----
  {
    const float* wg = in.p[13]; const float* bg = in.p[14];
    for (int u = tid; u < 512; u += 256) {
      int row = u >> 2, e = u & 3;
      const float* hr = hf + row * 260;
      float acc = bg[e];
      for (int d = 0; d < 256; ++d) acc += hr[d] * wg[d * 4 + e];
      glog[row * 4 + e] = acc;
    }
    __syncthreads();
    if (tid < 128) {
      float* lp = glog + tid * 4;
      float m = fmaxf(fmaxf(lp[0], lp[1]), fmaxf(lp[2], lp[3]));
      float e0 = __expf(lp[0] - m), e1 = __expf(lp[1] - m);
      float e2 = __expf(lp[2] - m), e3 = __expf(lp[3] - m);
      float inv = 1.f / (e0 + e1 + e2 + e3);
      gatev[tid * 4 + 0] = e0 * inv; gatev[tid * 4 + 1] = e1 * inv;
      gatev[tid * 4 + 2] = e2 * inv; gatev[tid * 4 + 3] = e3 * inv;
    }
  }
  __syncthreads();

  // ---- Phase H: MoE, expert loop; gated accumulate into hf ----
  bf16_t* hid = ub;  // [128][136], reuses qkv/ctx space
  {
    const float* b1 = in.p[16]; const float* b2 = in.p[18];
    for (int e = 0; e < 4; ++e) {
      {   // hid = relu(h @ w1[e] + b1[e]) : 8 nt -> 8 units, one per wave
        int nt2 = wid & 3, mh = wid >> 2;
        v8f c[4][2]; zero_acc(c);
        gemm_4x2(hb + mh * 64 * 264, 264,
                 ws + WS_W1 + e * 32768 + (nt2 * 2 + 0) * 16 * 256,
                 ws + WS_W1 + e * 32768 + (nt2 * 2 + 1) * 16 * 256, 256, 256, c, lane);
        #pragma unroll
        for (int n = 0; n < 2; ++n) {
          int col = (nt2 * 2 + n) * 16 + nloc;
          float bv = b1[e * 128 + col];
          #pragma unroll
          for (int mt = 0; mt < 4; ++mt) {
            #pragma unroll
            for (int i = 0; i < 8; ++i) {
              int row = (mh * 4 + mt) * 16 + half * 8 + i;
              hid[row * 136 + col] = (bf16_t)fmaxf(c[mt][n][i] + bv, 0.f);
            }
          }
        }
      }
      __syncthreads();
      for (int u = wid; u < 16; u += 8) {   // hf += gate[e] * (hid @ w2[e] + b2[e])
        int nt2 = u & 7, mh = u >> 3;
        v8f c[4][2]; zero_acc(c);
        gemm_4x2(hid + mh * 64 * 136, 136,
                 ws + WS_W2 + e * 32768 + (nt2 * 2 + 0) * 16 * 128,
                 ws + WS_W2 + e * 32768 + (nt2 * 2 + 1) * 16 * 128, 128, 128, c, lane);
        #pragma unroll
        for (int n = 0; n < 2; ++n) {
          int col = (nt2 * 2 + n) * 16 + nloc;
          float bv = b2[e * 256 + col];
          #pragma unroll
          for (int mt = 0; mt < 4; ++mt) {
            #pragma unroll
            for (int i = 0; i < 8; ++i) {
              int row = (mh * 4 + mt) * 16 + half * 8 + i;
              hf[row * 260 + col] += (c[mt][n][i] + bv) * gatev[row * 4 + e];
            }
          }
        }
      }
      __syncthreads();
    }
  }

  // ---- Phase I: LN2 ----
  layernorm_rows(hf, hb, in.p[19], in.p[20], tid, nullptr);
  __syncthreads();

  // ---- Phase J: conv, grouped by tap so B frags are reused across s_out ----
  for (int u = wid; u < 16; u += 8) {
    int nt2 = u & 7, mh = u >> 3;
    v8f c[4][2]; zero_acc(c);
    const int m = nloc;
    for (int t2 = 0; t2 < 5; ++t2) {
      int s0 = TAPSI[t2][mh * 4 + 0], s1 = TAPSI[t2][mh * 4 + 1];
      int s2 = TAPSI[t2][mh * 4 + 2], s3 = TAPSI[t2][mh * 4 + 3];
      if ((s0 | s1 | s2 | s3) < 0 && s0 < 0 && s1 < 0 && s2 < 0 && s3 < 0) continue;
      const bf16_t* B0 = ws + WS_CONV + (t2 * 2) * 65536 + (nt2 * 2 + 0) * 16 * 256;
      const bf16_t* B1 = B0 + 16 * 256;
      for (int k0 = 0; k0 < 256; k0 += 32) {
        union { v16bf v; uint4 q[2]; } b0, b1;
        const bf16_t* bp0 = B0 + m * 256 + k0 + half * 16;
        b0.q[0] = *(const uint4*)(bp0);
        b0.q[1] = *(const uint4*)(bp0 + 8);
        const bf16_t* bp1 = B1 + m * 256 + k0 + half * 16;
        b1.q[0] = *(const uint4*)(bp1);
        b1.q[1] = *(const uint4*)(bp1 + 8);
        #pragma unroll
        for (int sl = 0; sl < 4; ++sl) {
          int si = TAPSI[t2][mh * 4 + sl];
          if (si < 0) continue;
          union { v16bf v; uint4 q[2]; } a;
          const bf16_t* ap = hb + (si * 16 + m) * 264 + k0 + half * 16;
          a.q[0] = *(const uint4*)(ap);
          a.q[1] = *(const uint4*)(ap + 8);
          c[sl][0] = __builtin_amdgcn_wmma_f32_16x16x32_bf16(false, a.v, false, b0.v,
                                                             (short)0, c[sl][0], false, false);
          c[sl][1] = __builtin_amdgcn_wmma_f32_16x16x32_bf16(false, a.v, false, b1.v,
                                                             (short)0, c[sl][1], false, false);
        }
      }
    }
    #pragma unroll
    for (int n = 0; n < 2; ++n) {
      int col = (nt2 * 2 + n) * 16 + nloc;
      #pragma unroll
      for (int sl = 0; sl < 4; ++sl) {
        int so = mh * 4 + sl;
        float pb = pos_bias[so * 256 + col];
        #pragma unroll
        for (int i = 0; i < 8; ++i) {
          int row = so * 16 + half * 8 + i;
          hf[row * 260 + col] += c[sl][n][i] + pb;
        }
      }
    }
  }
  __syncthreads();

  // ---- Phase K: LN3 -> global output ----
  {
    int row = tid >> 1;
    int lb = row & 15, s = row >> 4;
    float* gptr = out + ((size_t)((b0 + lb) * 8 + s)) * 256;
    layernorm_rows(hf, hb, in.p[24], in.p[25], tid, gptr);
  }
}

// ---------------- weight-prep kernels ----------------
__global__ void kprep_embed(const float* we, unsigned short* dst) {
  int i = blockIdx.x * 256 + threadIdx.x;
  if (i >= 8192) return;
  int n = i >> 5, k = i & 31;
  float v = (k < 16) ? we[k * 256 + n] : 0.f;
  ((bf16_t*)dst)[i] = (bf16_t)v;
}

__global__ void kprep_mat(const float* src, unsigned short* dst, int K, int N) {
  int i = blockIdx.x * 256 + threadIdx.x;
  if (i >= N * K) return;
  int n = i / K, k = i - n * K;
  ((bf16_t*)dst)[i] = (bf16_t)src[k * N + n];   // [K][N] f32 -> [N][K] bf16
}

__global__ void kprep_posbias(const float* conv_w, const float* conv_b,
                              const float* pad_vec, float* pb) {
  int i = blockIdx.x * 256 + threadIdx.x;
  if (i >= 2048) return;
  int s = i >> 8, d = i & 255;
  const int PR[8] = {0,0,1,1,2,2,3,3};
  const int PC[8] = {1,3,0,2,1,3,0,2};
  int tok[16];
  #pragma unroll
  for (int j = 0; j < 16; ++j) tok[j] = 0;
  #pragma unroll
  for (int t = 0; t < 8; ++t) tok[PR[t] * 4 + PC[t]] = 1;
  float acc = conv_b[d];
  for (int dr = 0; dr < 3; ++dr)
    for (int dc = 0; dc < 3; ++dc) {
      int nr = PR[s] + dr - 1, nc = PC[s] + dc - 1;
      if (nr < 0 || nr > 3 || nc < 0 || nc > 3) continue;
      if (tok[nr * 4 + nc]) continue;               // token taps handled by WMMA edges
      const float* wt = conv_w + (size_t)(dr * 3 + dc) * 65536 + d;
      float a = 0.f;
      for (int ii = 0; ii < 256; ++ii) a += pad_vec[ii] * wt[ii * 256];
      acc += a;
    }
  pb[i] = acc;
}

// ---------------- host launch ----------------
static void prep_mat(const float* s, unsigned short* d, int K, int N, hipStream_t st) {
  kprep_mat<<<(N * K + 255) / 256, 256, 0, st>>>(s, d, K, N);
}

extern "C" void kernel_launch(void* const* d_in, const int* in_sizes, int n_in,
                              void* d_out, int out_size, void* d_ws, size_t ws_size,
                              hipStream_t stream) {
  const float* const* in = (const float* const*)d_in;
  unsigned short* ws = (unsigned short*)d_ws;
  float* pos_bias = (float*)((char*)d_ws + (size_t)WS_BF16_END * 2);

  kprep_embed<<<32, 256, 0, stream>>>(in[1], ws + WS_WEMB);
  prep_mat(in[3], ws + WS_WQ, 256, 64, stream);
  prep_mat(in[5], ws + WS_WK, 256, 64, stream);
  prep_mat(in[7], ws + WS_WV, 256, 64, stream);
  prep_mat(in[9], ws + WS_WO, 64, 256, stream);
  for (int e = 0; e < 4; ++e)
    prep_mat(in[15] + e * 32768, ws + WS_W1 + e * 32768, 256, 128, stream);
  for (int e = 0; e < 4; ++e)
    prep_mat(in[17] + e * 32768, ws + WS_W2 + e * 32768, 128, 256, stream);
  for (int t = 0; t < 9; ++t)
    prep_mat(in[22] + t * 65536, ws + WS_CONV + t * 65536, 256, 256, stream);
  kprep_posbias<<<8, 256, 0, stream>>>(in[22], in[23], in[21], pos_bias);

  P26 P;
  for (int i = 0; i < 26; ++i) P.p[i] = in[i];

  size_t smem = 278528;  // ~272KB of the 320KB WGP LDS
  hipFuncSetAttribute((const void*)fused_block,
                      hipFuncAttributeMaxDynamicSharedMemorySize, (int)smem);
  fused_block<<<16384 / 16, 256, smem, stream>>>(P, ws, pos_bias, (float*)d_out);
}
